// Linear_AB_I8_DE_F32_7146825580666
// MI455X (gfx1250) — 8-bit matrix operand order repaired and checked by reading, compiled, not run on hardware
//
#include <hip/hip_runtime.h>


#define ALPHA 0.01f

typedef __attribute__((ext_vector_type(8))) int v8i;

__device__ __forceinline__ unsigned int pack4(int4 v) {
  return ((unsigned)v.x & 0xFFu)
       | (((unsigned)v.y & 0xFFu) << 8)
       | (((unsigned)v.z & 0xFFu) << 16)
       | (((unsigned)v.w & 0xFFu) << 24);
}

__global__ __launch_bounds__(256) void pack_a_tiles(const int* __restrict__ src,
                                                    unsigned int* __restrict__ dst,
                                                    int K, int n_uints) {
  int id = blockIdx.x * blockDim.x + threadIdx.x;
  if (id >= n_uints) return;
  const int ktiles = K >> 6;
  int tile = id >> 8;
  int rem  = id & 255;
  int l = rem >> 3, j = rem & 7;
  int rt = tile / ktiles, kt = tile - rt * ktiles;
  int m = rt * 16 + (l & 15);
  int h = l >> 4;
  int k = kt * 64 + 16 * (j >> 1) + 8 * h + 4 * (j & 1);
  int4 v = *(const int4*)(src + (size_t)m * K + k);
  dst[id] = pack4(v);
}

__global__ __launch_bounds__(256) void pack_b_tiles(const int* __restrict__ src,
                                                    unsigned int* __restrict__ dst,
                                                    int K, int n_uints) {
  int id = blockIdx.x * blockDim.x + threadIdx.x;
  if (id >= n_uints) return;
  const int ktiles = K >> 6;
  int tile = id >> 8;
  int rem  = id & 255;
  int l = rem >> 3, j = rem & 7;
  int rt = tile / ktiles, kt = tile - rt * ktiles;
  int n = rt * 16 + (l & 15);
  int h = l >> 4;
  int k = kt * 64 + 16 * (j >> 1) + 8 * h + 4 * (j & 1);
  int4 v = *(const int4*)(src + (size_t)n * K + k);
  dst[id] = pack4(v);
}

__device__ __forceinline__ v8i load_frag(const char* ubase, int lane32) {
  const char* p = ubase + lane32;
  int4 lo = *(const int4*)(p);
  int4 hi = *(const int4*)(p + 16);
  v8i f;
  f[0] = lo.x; f[1] = lo.y; f[2] = lo.z; f[3] = lo.w;
  f[4] = hi.x; f[5] = hi.y; f[6] = hi.z; f[7] = hi.w;
  return f;
}

__device__ __forceinline__ v8i wmma_iu8(v8i a, v8i b, v8i c) {
  return __builtin_amdgcn_wmma_i32_16x16x64_iu8(
       true, a,  true, b, c,
       false,  false);
}

__device__ __forceinline__ void kstep(const char* ap, const char* bp,
                                      const char* apn, const char* bpn,
                                      size_t rowt, int lane32,
                                      v8i (&bu)[4], v8i (&bf)[4],
                                      v8i& a0, v8i& a1, v8i (&acc)[4][4]) {
  a1 = load_frag(ap + rowt, lane32);
  acc[0][0] = wmma_iu8(a0, bu[0], acc[0][0]);
  acc[0][1] = wmma_iu8(a0, bu[1], acc[0][1]);
  acc[0][2] = wmma_iu8(a0, bu[2], acc[0][2]);
  acc[0][3] = wmma_iu8(a0, bu[3], acc[0][3]);
  a0 = load_frag(ap + 2 * rowt, lane32);
  bf[0] = load_frag(bpn, lane32);
  bf[1] = load_frag(bpn + rowt, lane32);
  bf[2] = load_frag(bpn + 2 * rowt, lane32);
  bf[3] = load_frag(bpn + 3 * rowt, lane32);
  acc[1][0] = wmma_iu8(a1, bu[0], acc[1][0]);
  acc[1][1] = wmma_iu8(a1, bu[1], acc[1][1]);
  acc[1][2] = wmma_iu8(a1, bu[2], acc[1][2]);
  acc[1][3] = wmma_iu8(a1, bu[3], acc[1][3]);
  a1 = load_frag(ap + 3 * rowt, lane32);
  acc[2][0] = wmma_iu8(a0, bu[0], acc[2][0]);
  acc[2][1] = wmma_iu8(a0, bu[1], acc[2][1]);
  acc[2][2] = wmma_iu8(a0, bu[2], acc[2][2]);
  acc[2][3] = wmma_iu8(a0, bu[3], acc[2][3]);
  a0 = load_frag(apn, lane32);
  acc[3][0] = wmma_iu8(a1, bu[0], acc[3][0]);
  acc[3][1] = wmma_iu8(a1, bu[1], acc[3][1]);
  acc[3][2] = wmma_iu8(a1, bu[2], acc[3][2]);
  acc[3][3] = wmma_iu8(a1, bu[3], acc[3][3]);
}

template <int KTILES>
__global__ __launch_bounds__(256) void gemm_i8_wmma(const char* __restrict__ x8t,
                                                    const char* __restrict__ w8t,
                                                    const float* __restrict__ bias,
                                                    float* __restrict__ out,
                                                    int M, int N, int K) {
  const int lane = threadIdx.x & 31;
  const int wave = threadIdx.x >> 5;
  const int half = lane >> 4;
  const int l16  = lane & 15;
  const int lane32 = lane * 32;

  const int wm = wave & 1;
  const int wn = wave >> 1;
  const int bm = blockIdx.y * 128 + wm * 64;
  const int bn = blockIdx.x * 256 + wn * 64;

  const int ktiles = (KTILES > 0) ? KTILES : (K >> 6);
  const size_t rowt = (size_t)ktiles * 1024;

  const char* apt = x8t + (size_t)(bm >> 4) * rowt;
  const char* bpt = w8t + (size_t)(bn >> 4) * rowt;

  v8i acc[4][4];
#pragma unroll
  for (int mi = 0; mi < 4; ++mi)
#pragma unroll
    for (int ni = 0; ni < 4; ++ni)
      acc[mi][ni] = (v8i)(0);

  v8i bA[4], bB[4], a0, a1;
#pragma unroll
  for (int i = 0; i < 4; ++i) bA[i] = load_frag(bpt + i * rowt, lane32);
  a0 = load_frag(apt, lane32);

  if constexpr (KTILES > 0) {
#pragma unroll
    for (int kt = 0; kt < KTILES; kt += 2) {
      const char* ap0 = apt + (size_t)kt * 1024;
      const char* bp0 = bpt + (size_t)kt * 1024;
      const char* ap1 = ap0 + 1024;
      const char* bp1 = bp0 + 1024;
      const char* ap2 = (kt + 2 >= KTILES) ? ap1 : (ap1 + 1024);
      const char* bp2 = (kt + 2 >= KTILES) ? bp1 : (bp1 + 1024);
      kstep(ap0, bp0, ap1, bp1, rowt, lane32, bA, bB, a0, a1, acc);
      kstep(ap1, bp1, ap2, bp2, rowt, lane32, bB, bA, a0, a1, acc);
    }
  } else {
#pragma unroll 1
    for (int kt = 0; kt < ktiles; kt += 2) {
      const char* ap1 = apt + 1024;
      const char* bp1 = bpt + 1024;
      const bool last = (kt + 2 >= ktiles);
      const char* ap2 = last ? ap1 : (apt + 2048);
      const char* bp2 = last ? bp1 : (bpt + 2048);
      kstep(apt, bpt, ap1, bp1, rowt, lane32, bA, bB, a0, a1, acc);
      kstep(ap1, bp1, ap2, bp2, rowt, lane32, bB, bA, a0, a1, acc);
      apt = ap2;
      bpt = bp2;
    }
  }

#pragma unroll
  for (int mi = 0; mi < 4; ++mi) {
#pragma unroll
    for (int ni = 0; ni < 4; ++ni) {
      const int n = bn + ni * 16 + l16;
      const float bv = bias[n];
      const int mbase = bm + mi * 16 + 8 * half;
      float* o = out + (size_t)mbase * (size_t)N + n;
#pragma unroll
      for (int r = 0; r < 8; ++r)
        o[(size_t)r * N] = ALPHA * (float)acc[mi][ni][r] + bv;
    }
  }
}

extern "C" void kernel_launch(void* const* d_in, const int* in_sizes, int n_in,
                              void* d_out, int out_size, void* d_ws, size_t ws_size,
                              hipStream_t stream) {
  const int* x_i32 = (const int*)d_in[0];
  const int* w_i32 = (const int*)d_in[1];
  const float* bias = (const float*)d_in[2];
  float* out = (float*)d_out;

  const int N = in_sizes[2];
  const int K = in_sizes[1] / N;
  const int M = in_sizes[0] / K;

  char* x8t = (char*)d_ws;
  char* w8t = x8t + (size_t)M * (size_t)K;

  {
    int n_uints = (M * K) / 4;
    pack_a_tiles<<<(n_uints + 255) / 256, 256, 0, stream>>>(
        x_i32, (unsigned int*)x8t, K, n_uints);
  }
  {
    int n_uints = (int)(((long long)N * (long long)K) / 4);
    pack_b_tiles<<<(n_uints + 255) / 256, 256, 0, stream>>>(
        w_i32, (unsigned int*)w8t, K, n_uints);
  }

  dim3 grid(N / 256, M / 128);
  if (K == 4096) {
    gemm_i8_wmma<64><<<grid, 256, 0, stream>>>(x8t, w8t, bias, out, M, N, K);
  } else {
    gemm_i8_wmma<0><<<grid, 256, 0, stream>>>(x8t, w8t, bias, out, M, N, K);
  }
}
